// Attention_45148696216391
// MI455X (gfx1250) — compile-verified, matrix operand order checked by reading, not run on hardware
//
#include <hip/hip_runtime.h>
#include <hip/hip_bf16.h>

typedef __attribute__((ext_vector_type(16))) _Float16 v16h;
typedef __attribute__((ext_vector_type(8)))  _Float16 v8h;
typedef __attribute__((ext_vector_type(8)))  float    v8f;
typedef __attribute__((ext_vector_type(4)))  int      v4i;

// Problem constants
#define BQ   2
#define NSEQ 2048
#define DIM  1024
#define HEADS 16
#define DH   64
#define MROWS 4096   // B*N

// CDNA5 async global->LDS path (guarded; fallback = ds_store staging)
#if defined(__has_builtin)
# if __has_builtin(__builtin_amdgcn_global_load_async_to_lds_b128)
#  define HAVE_ASYNC 1
# endif
#endif
#ifndef HAVE_ASYNC
# define HAVE_ASYNC 0
#endif

#if HAVE_ASYNC
typedef __attribute__((address_space(1))) v4i gv4i;  // global 16B chunk
typedef __attribute__((address_space(3))) v4i lv4i;  // LDS 16B chunk
__device__ __forceinline__ gv4i* as_g(const void* p) {
  return (gv4i*)(unsigned long long)p;
}
__device__ __forceinline__ lv4i* as_l(void* p) {
  return (lv4i*)(unsigned int)(unsigned long long)p;  // low 32b = LDS offset
}
#endif

__device__ __forceinline__ v8f wmma_f16(v16h a, v16h b, v8f c) {
  // D = A(16x32 f16) * B(32x16 f16) + C(16x16 f32)
  return __builtin_amdgcn_wmma_f32_16x16x32_f16(
      /*neg_a=*/false, a, /*neg_b=*/false, b,
      /*c_mod=*/(short)0, c, /*reuse_a=*/false, /*reuse_b=*/false);
}

// Load a 16-half K-slice for one lane of an A- (or Bt-) fragment.
// Per ISA: lane<16 -> K = kbase+{0..7, 16..23}; lane>=16 -> K = kbase+{8..15, 24..31}
__device__ __forceinline__ v16h load_frag(const _Float16* __restrict__ rowPtr,
                                          int hi, int kbase) {
  union { v16h v; v8h h[2]; } u;
  u.h[0] = *(const v8h*)(rowPtr + kbase + hi * 8);
  u.h[1] = *(const v8h*)(rowPtr + kbase + 16 + hi * 8);
  return u.v;
}

__device__ __forceinline__ float rmax16(float x) {
#pragma unroll
  for (int m = 1; m < 16; m <<= 1) x = fmaxf(x, __shfl_xor(x, m, 32));
  return x;
}
__device__ __forceinline__ float rsum16(float x) {
#pragma unroll
  for (int m = 1; m < 16; m <<= 1) x += __shfl_xor(x, m, 32);
  return x;
}

__global__ void f32_to_f16_kernel(const float* __restrict__ in,
                                  _Float16* __restrict__ out, int n) {
  int i = blockIdx.x * blockDim.x + threadIdx.x;
  if (i < n) out[i] = (_Float16)in[i];
}

// Fused GEMM: C(M=4096 x N=1024) = A(4096x1024,f16 rowmajor) @ Bm^T
// Block = 256 threads = 8 waves; block tile 256(M) x 64(N); each wave 32x64.
// B tile (64 rows x 32 K) staged cooperatively in LDS (async-to-LDS on CDNA5).
// mode 0: Q epilogue  (L2 norm * q_scale * DH^-0.5 -> f16 (B,H,N,DH))
// mode 1: K epilogue  (L2 norm * k_scale           -> f16 (B,H,N,DH))
// mode 2: V epilogue  (transpose                   -> f16 (B,H,DH,N))
// mode 3: OUT epilogue(+bias                       -> f32 (B,N,DIM))
#define BROW 40  // padded LDS row stride in halves (80B: conflict-free frags)
__global__ __launch_bounds__(256)
void gemm_wmma_kernel(const _Float16* __restrict__ A,
                      const _Float16* __restrict__ Bm,
                      void* __restrict__ outp,
                      const float* __restrict__ aux, int mode) {
  __shared__ __attribute__((aligned(16))) _Float16 Bt[64 * BROW];

  const int tid  = threadIdx.x;
  const int lane = tid & 31;
  const int wv   = tid >> 5;
  const int hi   = lane >> 4;
  const int ln   = lane & 15;
  const int Mbase = blockIdx.x * 256 + wv * 32;  // wave's 32 output rows
  const int Nbase = blockIdx.y * 64;             // block's 64 output cols

  v8f c[2][4] = {};

  const _Float16* arow0 = A + (size_t)(Mbase + ln) * DIM;
  const _Float16* arow1 = arow0 + (size_t)16 * DIM;

  // Cooperative B-tile staging: each of the 256 threads moves one 16B chunk.
  const int srow   = tid >> 2;  // 0..63 : row of B tile
  const int schunk = tid & 3;   // 0..3  : 8-half chunk within the 32-K slice
  const _Float16* gsrc = Bm + (size_t)(Nbase + srow) * DIM + schunk * 8;
  _Float16* ldst = Bt + srow * BROW + schunk * 8;

  for (int k = 0; k < DIM; k += 32) {
#if HAVE_ASYNC
    __builtin_amdgcn_global_load_async_to_lds_b128(as_g(gsrc + k), as_l(ldst),
                                                   0, 0);
    asm volatile("s_wait_asynccnt 0" ::: "memory");
#else
    *(v8h*)ldst = *(const v8h*)(gsrc + k);
#endif
    __syncthreads();

    v16h a0 = load_frag(arow0, hi, k);
    v16h a1 = load_frag(arow1, hi, k);
#pragma unroll
    for (int t = 0; t < 4; ++t) {
      const _Float16* br = Bt + (t * 16 + ln) * BROW;
      union { v16h v; v8h h[2]; } u;
      u.h[0] = *(const v8h*)(br + hi * 8);
      u.h[1] = *(const v8h*)(br + 16 + hi * 8);
      c[0][t] = wmma_f16(a0, u.v, c[0][t]);
      c[1][t] = wmma_f16(a1, u.v, c[1][t]);
    }
    __syncthreads();
  }

  if (mode == 3) {  // output projection: f32 + bias -> d_out (B,N,DIM)
    float* out = (float*)outp;
#pragma unroll
    for (int rb = 0; rb < 2; ++rb) {
#pragma unroll
      for (int t = 0; t < 4; ++t) {
        int col = Nbase + t * 16 + ln;
        float bias = aux[col];
#pragma unroll
        for (int v = 0; v < 8; ++v) {
          int r = Mbase + rb * 16 + v + 8 * hi;
          out[(size_t)r * DIM + col] = c[rb][t][v] + bias;
        }
      }
    }
    return;
  }

  const int h = Nbase >> 6;  // one head per 64-col tile
  _Float16* out = (_Float16*)outp;

  if (mode == 2) {  // V: transpose into (B,H,DH,N)
#pragma unroll
    for (int rb = 0; rb < 2; ++rb) {
#pragma unroll
      for (int t = 0; t < 4; ++t) {
        int d = t * 16 + ln;
#pragma unroll
        for (int v = 0; v < 8; ++v) {
          int r = Mbase + rb * 16 + v + 8 * hi;
          int b = r >> 11, n = r & (NSEQ - 1);
          size_t off = (((size_t)(b * HEADS + h)) * DH + d) * NSEQ + n;
          out[off] = (_Float16)c[rb][t][v];
        }
      }
    }
    return;
  }

  // mode 0/1: per-row L2 normalize over the 64 head dims, then scale.
  const float extra = (mode == 0) ? 0.125f : 1.0f;  // fold DH^-0.5 into Q
#pragma unroll
  for (int rb = 0; rb < 2; ++rb) {
    float inv[8];
#pragma unroll
    for (int v = 0; v < 8; ++v) {
      float s = c[rb][0][v] * c[rb][0][v] + c[rb][1][v] * c[rb][1][v] +
                c[rb][2][v] * c[rb][2][v] + c[rb][3][v] * c[rb][3][v];
      s = rsum16(s);
      inv[v] = extra / fmaxf(sqrtf(s), 1e-12f);
    }
#pragma unroll
    for (int t = 0; t < 4; ++t) {
      int d = t * 16 + ln;
      float sc = aux[h * DH + d];  // q_scale/k_scale is (H,1,DH)
#pragma unroll
      for (int v = 0; v < 8; ++v) {
        int r = Mbase + rb * 16 + v + 8 * hi;
        int b = r >> 11, n = r & (NSEQ - 1);
        size_t off = (((size_t)(b * HEADS + h)) * NSEQ + n) * DH + d;
        out[off] = (_Float16)(c[rb][t][v] * inv[v] * sc);
      }
    }
  }
}

// Flash attention: one block = 8 waves, each wave owns 16 query rows.
// Q,K in (B,H,N,DH) f16 (scale pre-folded), V in (B,H,DH,N) f16.
// Output Ob in (B,N,H*DH) f16 for the final projection.
__global__ __launch_bounds__(256)
void attn_wmma_kernel(const _Float16* __restrict__ Qb,
                      const _Float16* __restrict__ Kb,
                      const _Float16* __restrict__ Vb,
                      _Float16* __restrict__ Ob) {
  __shared__ __attribute__((aligned(16))) _Float16 ldsP[8 * 16 * 32];

  const int lane = threadIdx.x & 31;
  const int wv   = threadIdx.x >> 5;
  const int hi   = lane >> 4;
  const int ln   = lane & 15;
  const int bh   = blockIdx.x >> 4;   // 0..31 (b*H + h)
  const int qt   = blockIdx.x & 15;
  const int qbase = qt * 128 + wv * 16;

  const _Float16* qp = Qb + (size_t)bh * NSEQ * DH;
  const _Float16* kp = Kb + (size_t)bh * NSEQ * DH;
  const _Float16* vp = Vb + (size_t)bh * DH * NSEQ;

  const _Float16* qrow = qp + (size_t)(qbase + ln) * DH;
  v16h aq0 = load_frag(qrow, hi, 0);    // dh 0..31
  v16h aq1 = load_frag(qrow, hi, 32);   // dh 32..63

  float m[8], l[8];
#pragma unroll
  for (int v = 0; v < 8; ++v) { m[v] = -3.0e38f; l[v] = 0.0f; }
  v8f o[4] = {{0}, {0}, {0}, {0}};

  _Float16* myP = ldsP + wv * 512;  // private 16x32 f16 staging tile

  for (int kb = 0; kb < NSEQ; kb += 32) {
    // S tiles: 16 queries x 32 keys (two 16x16 WMMA outputs), K-dim = DH
    const _Float16* krow0 = kp + (size_t)(kb + ln) * DH;
    const _Float16* krow1 = krow0 + 16 * DH;
    v8f s0 = {0}, s1 = {0};
    s0 = wmma_f16(aq0, load_frag(krow0, hi, 0),  s0);
    s0 = wmma_f16(aq1, load_frag(krow0, hi, 32), s0);
    s1 = wmma_f16(aq0, load_frag(krow1, hi, 0),  s1);
    s1 = wmma_f16(aq1, load_frag(krow1, hi, 32), s1);

    // Online softmax (scale already folded into Q)
    float alpha[8];
#pragma unroll
    for (int v = 0; v < 8; ++v) {
      float sm = rmax16(fmaxf(s0[v], s1[v]));
      float nm = fmaxf(m[v], sm);
      alpha[v] = __expf(m[v] - nm);
      m[v] = nm;
      float p0 = __expf(s0[v] - nm);
      float p1 = __expf(s1[v] - nm);
      s0[v] = p0; s1[v] = p1;
      l[v] = l[v] * alpha[v] + rsum16(p0 + p1);
    }
#pragma unroll
    for (int t = 0; t < 4; ++t)
#pragma unroll
      for (int v = 0; v < 8; ++v) o[t][v] *= alpha[v];

    // Stage P (C layout) -> LDS -> reload in A-fragment layout
#pragma unroll
    for (int v = 0; v < 8; ++v) {
      int r = v + 8 * hi;
      myP[r * 32 + ln]      = (_Float16)s0[v];
      myP[r * 32 + 16 + ln] = (_Float16)s1[v];
    }
    asm volatile("s_wait_dscnt 0" ::: "memory");
    v16h ap;
    {
      const _Float16* prow = myP + ln * 32;
      union { v16h vv; v8h h[2]; } u;
      u.h[0] = *(const v8h*)(prow + hi * 8);
      u.h[1] = *(const v8h*)(prow + 16 + hi * 8);
      ap = u.vv;
    }

    // O(16x64) += P(16x32) * V(32x64); V^T rows are contiguous in Vb
#pragma unroll
    for (int t = 0; t < 4; ++t) {
      const _Float16* vrow = vp + (size_t)(t * 16 + ln) * NSEQ;
      v16h bv = load_frag(vrow, hi, kb);
      o[t] = wmma_f16(ap, bv, o[t]);
    }
  }

  // Normalize by row sums and write (B, N, H*DH)
  const int b = bh >> 4, h = bh & (HEADS - 1);
#pragma unroll
  for (int v = 0; v < 8; ++v) {
    float inv = 1.0f / fmaxf(l[v], 1e-30f);
#pragma unroll
    for (int t = 0; t < 4; ++t) o[t][v] *= inv;
  }
#pragma unroll
  for (int t = 0; t < 4; ++t) {
#pragma unroll
    for (int v = 0; v < 8; ++v) {
      int n = qbase + v + 8 * hi;
      size_t off = ((size_t)(b * NSEQ + n)) * (HEADS * DH) + h * DH + t * 16 + ln;
      Ob[off] = (_Float16)o[t][v];
    }
  }
}

extern "C" void kernel_launch(void* const* d_in, const int* in_sizes, int n_in,
                              void* d_out, int out_size, void* d_ws, size_t ws_size,
                              hipStream_t stream) {
  const float* x       = (const float*)d_in[0];
  const float* Wq      = (const float*)d_in[1];
  const float* Wk      = (const float*)d_in[2];
  const float* Wv      = (const float*)d_in[3];
  const float* Wo      = (const float*)d_in[4];
  const float* bo      = (const float*)d_in[5];
  const float* q_scale = (const float*)d_in[6];
  const float* k_scale = (const float*)d_in[7];
  float* out = (float*)d_out;

  // Workspace partition (halves): 48 MB total
  _Float16* ws  = (_Float16*)d_ws;
  _Float16* xh  = ws;                    // 4096*1024
  _Float16* Wqh = xh  + (size_t)MROWS * DIM;
  _Float16* Wkh = Wqh + (size_t)DIM * DIM;
  _Float16* Wvh = Wkh + (size_t)DIM * DIM;
  _Float16* Woh = Wvh + (size_t)DIM * DIM;
  _Float16* Qb  = Woh + (size_t)DIM * DIM;   // (B,H,N,DH)
  _Float16* Kb  = Qb  + (size_t)MROWS * DIM;
  _Float16* Vb  = Kb  + (size_t)MROWS * DIM; // (B,H,DH,N)
  _Float16* Ob  = Vb  + (size_t)MROWS * DIM; // (B,N,H*DH)

  const int nx = MROWS * DIM;   // 4194304
  const int nw = DIM * DIM;     // 1048576
  f32_to_f16_kernel<<<(nx + 255) / 256, 256, 0, stream>>>(x,  xh,  nx);
  f32_to_f16_kernel<<<(nw + 255) / 256, 256, 0, stream>>>(Wq, Wqh, nw);
  f32_to_f16_kernel<<<(nw + 255) / 256, 256, 0, stream>>>(Wk, Wkh, nw);
  f32_to_f16_kernel<<<(nw + 255) / 256, 256, 0, stream>>>(Wv, Wvh, nw);
  f32_to_f16_kernel<<<(nw + 255) / 256, 256, 0, stream>>>(Wo, Woh, nw);

  dim3 g(MROWS / 256, DIM / 64);  // (16,16)
  gemm_wmma_kernel<<<g, 256, 0, stream>>>(xh, Wqh, Qb, q_scale, 0);
  gemm_wmma_kernel<<<g, 256, 0, stream>>>(xh, Wkh, Kb, k_scale, 1);
  gemm_wmma_kernel<<<g, 256, 0, stream>>>(xh, Wvh, Vb, nullptr, 2);

  attn_wmma_kernel<<<BQ * HEADS * (NSEQ / 128), 256, 0, stream>>>(Qb, Kb, Vb, Ob);

  gemm_wmma_kernel<<<g, 256, 0, stream>>>(Ob, Woh, out, bo, 3);
}
